// JPGFN_21414706938559
// MI455X (gfx1250) — matrix operand order repaired and checked by reading, not compiled, not run on hardware
//
#include <hip/hip_runtime.h>
#include <hip/hip_bf16.h>

// ---------------- problem constants (from reference) ----------------
#define NN   50000
#define DD   64
#define HH   64
#define OO   32
#define EE   800000
#define SPL  5
#define KA   (DD*SPL)   // 320
#define DEPTH 3

typedef __attribute__((ext_vector_type(16))) __bf16 v16bf;
typedef __attribute__((ext_vector_type(8)))  float  v8f;

// ---------------- WMMA helpers (CDNA5 16x16x32 bf16, f32 accum) ------
__device__ __forceinline__ v8f wmma_bf16(v16bf a, v16bf b, v8f c) {
  return __builtin_amdgcn_wmma_f32_16x16x32_bf16(
      /*neg_a=*/false, a, /*neg_b=*/false, b,
      /*c_mod=*/(short)0, c, /*reuse_a=*/false, /*reuse_b=*/false);
}

// A-matrix 16x32 bf16 fragment from row-major LDS tile (lda elements/row).
// lane l -> row m=l&15, half h=l>>4; elem i -> K = h*16+i.
// Both 8-element runs are contiguous -> lowers to 2x ds_load_b128.
__device__ __forceinline__ v16bf load_a_frag(const __bf16* A, int lda, int k0, int lane) {
  int m = lane & 15, hh = lane >> 4;
  const __bf16* p = A + m * lda + k0;
  v16bf a;
#pragma unroll
  for (int i = 0; i < 16; ++i) {
    int k = hh * 16 + i;
    a[i] = p[k];
  }
  return a;
}

// B fragments are pre-swizzled to fragment-major: tile t, lane l owns 16
// contiguous bf16 (32 bytes) -> a single v16bf LDS load (2x ds_load_b128).
__device__ __forceinline__ v16bf load_b_swz(const __bf16* B, int tile, int lane) {
  return *reinterpret_cast<const v16bf*>(B + (tile * 32 + lane) * 16);
}

// Swizzled destination index for B element (k, n) with `ntiles` 16-col tiles:
// lane l = ((k>>4)&1)*16 + (n&15), elem i = k&15, tile = (k>>5)*ntiles + (n>>4).
__device__ __forceinline__ int bswz(int k, int n, int ntiles) {
  int tile = (k >> 5) * ntiles + (n >> 4);
  int lane = ((k >> 4) & 1) * 16 + (n & 15);
  return (tile * 32 + lane) * 16 + (k & 15);
}

// ---------------- BatchNorm statistics -------------------------------
__global__ void k_bn_stats(const float* __restrict__ x, float* __restrict__ sums) {
  __shared__ float ls[DD], lss[DD];
  int t = threadIdx.x;
  if (t < DD) { ls[t] = 0.f; lss[t] = 0.f; }
  __syncthreads();
  int total = NN * DD;
  for (int i = blockIdx.x * blockDim.x + t; i < total; i += gridDim.x * blockDim.x) {
    float v = x[i];
    int d = i & (DD - 1);
    atomicAdd(&ls[d], v);
    atomicAdd(&lss[d], v * v);
  }
  __syncthreads();
  if (t < DD) {
    atomicAdd(&sums[t], ls[t]);
    atomicAdd(&sums[DD + t], lss[t]);
  }
}

__global__ void k_bn_final(const float* __restrict__ sums,
                           const float* __restrict__ gamma,
                           const float* __restrict__ beta,
                           float* __restrict__ scl, float* __restrict__ shf) {
  int d = threadIdx.x;
  if (d < DD) {
    float mean = sums[d] * (1.0f / (float)NN);
    float var  = sums[DD + d] * (1.0f / (float)NN) - mean * mean;
    float istd = rsqrtf(var + 1e-5f);
    float s = istd * gamma[d];
    scl[d] = s;
    shf[d] = beta[d] - mean * s;
  }
}

// ---------------- degrees + norms ------------------------------------
__global__ void k_deg(const int* __restrict__ row, const int* __restrict__ col,
                      float* __restrict__ degR, float* __restrict__ degC) {
  for (int e = blockIdx.x * blockDim.x + threadIdx.x; e < EE; e += gridDim.x * blockDim.x) {
    unsafeAtomicAdd(&degR[row[e]], 1.0f);
    unsafeAtomicAdd(&degC[col[e]], 1.0f);
  }
}

__global__ void k_norms(const float* __restrict__ degR, const float* __restrict__ degC,
                        float* __restrict__ dis, float* __restrict__ odn,
                        float* __restrict__ idn) {
  int n = blockIdx.x * blockDim.x + threadIdx.x;
  if (n < NN) {
    float dr = degR[n], dc = degC[n];
    dis[n] = (dr > 0.f) ? rsqrtf(dr) : 0.f;
    odn[n] = rsqrtf(fmaxf(dr, 1.f));
    idn[n] = rsqrtf(fmaxf(dc, 1.f));
  }
}

// ---------------- FSTNN pre-contraction (writes swizzled bf16) -------
// M[ds, h] = sum_k fst_w2[ds, k] * fst_cw[(d*H+k), h]
__global__ void k_precM(const float* __restrict__ w2, const float* __restrict__ cw,
                        __bf16* __restrict__ Mb) {
  int ds = blockIdx.x;        // 0..KA-1  (= d*SPL+s)  == K index
  int h  = threadIdx.x;       // 0..HH-1               == N index
  int d  = ds / SPL;
  float acc = 0.f;
  for (int k = 0; k < HH; ++k)
    acc += w2[ds * HH + k] * cw[(d * HH + k) * HH + h];
  Mb[bswz(ds, h, HH / 16)] = (__bf16)acc;
}

// biasH[h] = cb[h] + sum_{dk} b2[dk] * cw[dk, h]
__global__ void k_biasH(const float* __restrict__ b2, const float* __restrict__ cw,
                        const float* __restrict__ cb, float* __restrict__ biasH) {
  int h = threadIdx.x;
  float acc = cb[h];
  for (int dk = 0; dk < DD * HH; ++dk)
    acc += b2[dk] * cw[dk * HH + h];
  biasH[h] = acc;
}

// ---------------- swizzle small weight matrices to bf16 --------------
// dst layout: [w1 swz (4096)][w2 swz (2048)][gcn_cw swz (4096)]
__global__ void k_swzW(const float* __restrict__ w1, const float* __restrict__ w2,
                       const float* __restrict__ gcw, __bf16* __restrict__ dst) {
  int t = blockIdx.x * blockDim.x + threadIdx.x;
  if (t < HH * HH) {                       // w1: 64x64, ntiles=4
    int k = t / HH, n = t % HH;
    dst[bswz(k, n, 4)] = (__bf16)w1[t];
  } else if (t < HH * HH + HH * OO) {      // w2: 64x32, ntiles=2
    int u = t - HH * HH;
    int k = u / OO, n = u % OO;
    dst[HH * HH + bswz(k, n, 2)] = (__bf16)w2[u];
  } else if (t < HH * HH + HH * OO + DD * DD) {  // gcn_cw: 64x64, ntiles=4
    int u = t - HH * HH - HH * OO;
    int k = u / DD, n = u % DD;
    dst[HH * HH + HH * OO + bswz(k, n, 4)] = (__bf16)gcw[u];
  }
}

// ---------------- Jacobi scalar coefficients + softmax weights -------
__global__ void k_scalars(const float* __restrict__ alphas,
                          const float* __restrict__ combw,
                          float* __restrict__ coefs, float* __restrict__ wsm) {
  int t = threadIdx.x;
  if (t == 0) {
    float al[DEPTH + 1];
    for (int i = 0; i <= DEPTH; ++i) al[i] = 1.5f * tanhf(alphas[i]);
    const float AJ = -1.f, BJ = 2.f, LJ = -1.f, RJ = 1.f;
    float c1 = (AJ - BJ) * 0.5f - (AJ + BJ + 2.f) * 0.5f * (LJ + RJ) / (RJ - LJ);
    float c2 = (AJ + BJ + 2.f) / (RJ - LJ);
    coefs[0] = al[0] * c1;
    coefs[1] = al[0] * c2;
    for (int L = 2; L <= DEPTH; ++L) {
      float Lf = (float)L;
      float coef_l     = 2.f * Lf * (Lf + AJ + BJ) * (2.f * Lf - 2.f + AJ + BJ);
      float coef_lm1_1 = (2.f * Lf + AJ + BJ - 1.f) * (2.f * Lf + AJ + BJ) * (2.f * Lf + AJ + BJ - 2.f);
      float coef_lm1_2 = (2.f * Lf + AJ + BJ - 1.f) * (AJ * AJ - BJ * BJ);
      float coef_lm2   = 2.f * (Lf - 1.f + AJ) * (Lf - 1.f + BJ) * (2.f * Lf + AJ + BJ);
      float tmp1 = al[L - 1] * (coef_lm1_1 / coef_l);
      float tmp2 = al[L - 1] * (coef_lm1_2 / coef_l);
      float tmp3 = al[L - 1] * al[L - 2] * (coef_lm2 / coef_l);
      float tmp1_2 = tmp1 * (2.f / (RJ - LJ));
      float tmp2_2 = tmp1 * ((RJ + LJ) / (RJ - LJ)) + tmp2;
      int o = 2 + (L - 2) * 3;
      coefs[o] = tmp1_2; coefs[o + 1] = tmp2_2; coefs[o + 2] = tmp3;
    }
  }
  if (t < HH) {
    float m = combw[t];
    for (int L = 1; L <= DEPTH; ++L) m = fmaxf(m, combw[L * HH + t]);
    float e[DEPTH + 1], s = 0.f;
    for (int L = 0; L <= DEPTH; ++L) { e[L] = __expf(combw[L * HH + t] - m); s += e[L]; }
    float inv = 1.f / s;
    for (int L = 0; L <= DEPTH; ++L) wsm[L * HH + t] = e[L] * inv;
  }
}

// ---------------- h_emb = f1(xn) @ M + biasH  (WMMA) -----------------
__global__ __launch_bounds__(64) void k_hemb(const float* __restrict__ x,
    const float* __restrict__ scale, const float* __restrict__ shift,
    const float* __restrict__ w1, const float* __restrict__ b1,
    const __bf16* __restrict__ Mb, const float* __restrict__ biasH,
    float* __restrict__ hemb) {
  __shared__ __align__(16) __bf16 Bsh[KA * HH];       // 40960 B (swizzled)
  __shared__ __align__(16) __bf16 Ash[2][16 * KA];    // 20480 B
  __shared__ float w1s[KA], b1s[KA], scs[DD], shs[DD], bhs[HH];
  int tid = threadIdx.x, wave = tid >> 5, lane = tid & 31;

  {  // vectorized copy of pre-swizzled B: 2560 uint4
    const uint4* src = reinterpret_cast<const uint4*>(Mb);
    uint4* dst = reinterpret_cast<uint4*>(Bsh);
    for (int i = tid; i < (KA * HH) / 8; i += 64) dst[i] = src[i];
  }
  for (int i = tid; i < KA; i += 64) { w1s[i] = w1[i]; b1s[i] = b1[i]; }
  if (tid < DD) { scs[tid] = scale[tid]; shs[tid] = shift[tid]; }
  if (tid < HH) bhs[tid] = biasH[tid];

  int rowbase = blockIdx.x * 32 + wave * 16;
  __bf16* A = Ash[wave];
  for (int p = lane; p < 16 * DD; p += 32) {
    int m = p >> 6, d = p & (DD - 1);
    int grow = rowbase + m;
    float xv = 0.f;
    if (grow < NN) xv = x[grow * DD + d] * scs[d] + shs[d];
#pragma unroll
    for (int s = 0; s < SPL; ++s) {
      float f = xv * w1s[d * SPL + s] + b1s[d * SPL + s];
      f = (f > 0.f) ? f : 0.f;
      if (grow >= NN) f = 0.f;
      A[m * KA + d * SPL + s] = (__bf16)f;
    }
  }
  __syncthreads();

  int n = lane & 15, hh = lane >> 4;
  v8f acc[4] = {};
#pragma unroll
  for (int kt = 0; kt < KA / 32; ++kt) {
    v16bf a = load_a_frag(A, KA, kt * 32, lane);
#pragma unroll
    for (int nt = 0; nt < 4; ++nt) {
      v16bf b = load_b_swz(Bsh, kt * 4 + nt, lane);
      acc[nt] = wmma_bf16(a, b, acc[nt]);
    }
  }
#pragma unroll
  for (int nt = 0; nt < 4; ++nt)
#pragma unroll
    for (int r = 0; r < 8; ++r) {
      int grow = rowbase + r + 8 * hh;
      if (grow < NN) hemb[grow * HH + nt * 16 + n] = acc[nt][r] + bhs[nt * 16 + n];
    }
}

// ---------------- SpMM: out[row] += dis[row]*dis[col] * x[col] --------
__global__ void k_spmm_sym(const int* __restrict__ row, const int* __restrict__ col,
                           const float* __restrict__ dis, const float* __restrict__ x,
                           float* __restrict__ out) {
  int total = EE * 16;
  for (int idx = blockIdx.x * blockDim.x + threadIdx.x; idx < total;
       idx += gridDim.x * blockDim.x) {
    int e = idx >> 4, q = idx & 15;
    int r = row[e], c = col[e];
    float v = dis[r] * dis[c];
    float4 xv = *reinterpret_cast<const float4*>(x + c * HH + q * 4);
    float* o = out + r * HH + q * 4;
    unsafeAtomicAdd(o + 0, v * xv.x);
    unsafeAtomicAdd(o + 1, v * xv.y);
    unsafeAtomicAdd(o + 2, v * xv.z);
    unsafeAtomicAdd(o + 3, v * xv.w);
  }
}

// ---------------- SpMM: out[col] += odn[row]*idn[col] * x[row] --------
__global__ void k_spmm_gcn(const int* __restrict__ row, const int* __restrict__ col,
                           const float* __restrict__ odn, const float* __restrict__ idn,
                           const float* __restrict__ x, float* __restrict__ out) {
  int total = EE * 16;
  for (int idx = blockIdx.x * blockDim.x + threadIdx.x; idx < total;
       idx += gridDim.x * blockDim.x) {
    int e = idx >> 4, q = idx & 15;
    int r = row[e], c = col[e];
    float v = odn[r] * idn[c];
    float4 xv = *reinterpret_cast<const float4*>(x + r * DD + q * 4);
    float* o = out + c * DD + q * 4;
    unsafeAtomicAdd(o + 0, v * xv.x);
    unsafeAtomicAdd(o + 1, v * xv.y);
    unsafeAtomicAdd(o + 2, v * xv.z);
    unsafeAtomicAdd(o + 3, v * xv.w);
  }
}

// ---------------- Jacobi elementwise steps ---------------------------
__global__ void k_jstep1(const float* __restrict__ xs0, const float* __restrict__ t,
                         const float* __restrict__ coefs, float* __restrict__ xs1) {
  int i = blockIdx.x * blockDim.x + threadIdx.x;
  if (i < NN * HH) xs1[i] = coefs[0] * xs0[i] + coefs[1] * t[i];
}

__global__ void k_jstepL(const float* __restrict__ t, const float* __restrict__ xm1,
                         const float* __restrict__ xm2, const float* __restrict__ coefs,
                         int off, float* __restrict__ out) {
  int i = blockIdx.x * blockDim.x + threadIdx.x;
  if (i < NN * HH)
    out[i] = coefs[off] * t[i] - coefs[off + 1] * xm1[i] - coefs[off + 2] * xm2[i];
}

__global__ void k_comb(const float* __restrict__ x0, const float* __restrict__ x1,
                       const float* __restrict__ x2, const float* __restrict__ x3,
                       const float* __restrict__ wsm, float* __restrict__ hc) {
  int i = blockIdx.x * blockDim.x + threadIdx.x;
  if (i < NN * HH) {
    int h = i & (HH - 1);
    hc[i] = wsm[h] * x0[i] + wsm[HH + h] * x1[i] + wsm[2 * HH + h] * x2[i] +
            wsm[3 * HH + h] * x3[i];
  }
}

// ---------------- output MLP (two fused WMMA GEMMs) ------------------
__global__ __launch_bounds__(64) void k_mlp(const float* __restrict__ hc,
    const __bf16* __restrict__ Wswz,   // [w1 swz 4096][w2 swz 2048]
    const float* __restrict__ b1, const float* __restrict__ b2,
    float* __restrict__ hout) {
  __shared__ __align__(16) __bf16 W1s[HH * HH];
  __shared__ __align__(16) __bf16 W2s[HH * OO];
  __shared__ __align__(16) __bf16 Ash[2][16 * HH];
  __shared__ __align__(16) __bf16 Hsh[2][16 * HH];
  __shared__ float b1s[HH], b2s[OO];
  int tid = threadIdx.x, wave = tid >> 5, lane = tid & 31;
  {
    const uint4* src = reinterpret_cast<const uint4*>(Wswz);
    uint4* d1 = reinterpret_cast<uint4*>(W1s);
    uint4* d2 = reinterpret_cast<uint4*>(W2s);
    for (int i = tid; i < (HH * HH) / 8; i += 64) d1[i] = src[i];
    for (int i = tid; i < (HH * OO) / 8; i += 64) d2[i] = src[(HH * HH) / 8 + i];
  }
  if (tid < HH) b1s[tid] = b1[tid];
  if (tid < OO) b2s[tid] = b2[tid];

  int rowbase = blockIdx.x * 32 + wave * 16;
  __bf16* A = Ash[wave];
  for (int p = lane; p < 16 * HH; p += 32) {
    int m = p >> 6, hcol = p & (HH - 1);
    int grow = rowbase + m;
    A[p] = (__bf16)((grow < NN) ? hc[grow * HH + hcol] : 0.f);
  }
  __syncthreads();

  __bf16* Hd = Hsh[wave];
  int n = lane & 15, hh = lane >> 4;
  {
    v8f acc[4] = {};
#pragma unroll
    for (int kt = 0; kt < 2; ++kt) {
      v16bf a = load_a_frag(A, HH, kt * 32, lane);
#pragma unroll
      for (int nt = 0; nt < 4; ++nt) {
        v16bf b = load_b_swz(W1s, kt * 4 + nt, lane);
        acc[nt] = wmma_bf16(a, b, acc[nt]);
      }
    }
#pragma unroll
    for (int nt = 0; nt < 4; ++nt)
#pragma unroll
      for (int r = 0; r < 8; ++r) {
        int m = r + 8 * hh;
        float v = acc[nt][r] + b1s[nt * 16 + n];
        Hd[m * HH + nt * 16 + n] = (__bf16)((v > 0.f) ? v : 0.f);
      }
  }
  __syncthreads();
  {
    v8f acc[2] = {};
#pragma unroll
    for (int kt = 0; kt < 2; ++kt) {
      v16bf a = load_a_frag(Hd, HH, kt * 32, lane);
#pragma unroll
      for (int nt = 0; nt < 2; ++nt) {
        v16bf b = load_b_swz(W2s, kt * 2 + nt, lane);
        acc[nt] = wmma_bf16(a, b, acc[nt]);
      }
    }
#pragma unroll
    for (int nt = 0; nt < 2; ++nt)
#pragma unroll
      for (int r = 0; r < 8; ++r) {
        int grow = rowbase + r + 8 * hh;
        if (grow < NN) hout[grow * OO + nt * 16 + n] = acc[nt][r] + b2s[nt * 16 + n];
      }
  }
}

// ---------------- GCN per-dim spline ---------------------------------
__global__ void k_g2(const float* __restrict__ ax, const float* __restrict__ gw1,
                     const float* __restrict__ gb1, const float* __restrict__ gw2,
                     float* __restrict__ g2) {
  int i = blockIdx.x * blockDim.x + threadIdx.x;
  if (i < NN * DD) {
    int d = i & (DD - 1);
    float a = ax[i], acc = 0.f;
#pragma unroll
    for (int s = 0; s < SPL; ++s) {
      float f = a * gw1[d * SPL + s] + gb1[d * SPL + s];
      acc += ((f > 0.f) ? f : 0.f) * gw2[d * SPL + s];
    }
    g2[i] = acc;
  }
}

// ---------------- final GCN projection (WMMA) ------------------------
__global__ __launch_bounds__(64) void k_gcn_out(const float* __restrict__ t5,
    const float* __restrict__ gb2, const __bf16* __restrict__ Wswz,
    const float* __restrict__ cb, float* __restrict__ emb) {
  __shared__ __align__(16) __bf16 Ws[DD * DD];
  __shared__ __align__(16) __bf16 Ash[2][16 * DD];
  __shared__ float gb2s[DD], cbs[DD];
  int tid = threadIdx.x, wave = tid >> 5, lane = tid & 31;
  {
    const uint4* src = reinterpret_cast<const uint4*>(Wswz);
    uint4* dst = reinterpret_cast<uint4*>(Ws);
    for (int i = tid; i < (DD * DD) / 8; i += 64) dst[i] = src[i];
  }
  if (tid < DD) { gb2s[tid] = gb2[tid]; cbs[tid] = cb[tid]; }
  __syncthreads();

  int rowbase = blockIdx.x * 32 + wave * 16;
  __bf16* A = Ash[wave];
  for (int p = lane; p < 16 * DD; p += 32) {
    int m = p >> 6, d = p & (DD - 1);
    int grow = rowbase + m;
    A[p] = (__bf16)((grow < NN) ? (t5[grow * DD + d] + gb2s[d]) : 0.f);
  }
  __syncthreads();

  int n = lane & 15, hh = lane >> 4;
  v8f acc[4] = {};
#pragma unroll
  for (int kt = 0; kt < 2; ++kt) {
    v16bf a = load_a_frag(A, DD, kt * 32, lane);
#pragma unroll
    for (int nt = 0; nt < 4; ++nt) {
      v16bf b = load_b_swz(Ws, kt * 4 + nt, lane);
      acc[nt] = wmma_bf16(a, b, acc[nt]);
    }
  }
#pragma unroll
  for (int nt = 0; nt < 4; ++nt)
#pragma unroll
    for (int r = 0; r < 8; ++r) {
      int grow = rowbase + r + 8 * hh;
      if (grow < NN) emb[grow * DD + nt * 16 + n] = acc[nt][r] + cbs[nt * 16 + n];
    }
}

// ---------------- host-side orchestration ----------------------------
extern "C" void kernel_launch(void* const* d_in, const int* in_sizes, int n_in,
                              void* d_out, int out_size, void* d_ws, size_t ws_size,
                              hipStream_t stream) {
  (void)in_sizes; (void)n_in; (void)out_size; (void)ws_size;
  const float* in_feat = (const float*)d_in[0];
  const int*   ei      = (const int*)d_in[1];
  const float* bn_g    = (const float*)d_in[2];
  const float* bn_b    = (const float*)d_in[3];
  const float* fst_w1  = (const float*)d_in[4];
  const float* fst_b1  = (const float*)d_in[5];
  const float* fst_w2  = (const float*)d_in[6];
  const float* fst_b2  = (const float*)d_in[7];
  const float* fst_cw  = (const float*)d_in[8];
  const float* fst_cb  = (const float*)d_in[9];
  const float* alphas  = (const float*)d_in[10];
  const float* combw   = (const float*)d_in[11];
  const float* out_w1  = (const float*)d_in[12];
  const float* out_b1  = (const float*)d_in[13];
  const float* out_w2  = (const float*)d_in[14];
  const float* out_b2  = (const float*)d_in[15];
  const float* gcn_w1  = (const float*)d_in[16];
  const float* gcn_b1  = (const float*)d_in[17];
  const float* gcn_w2  = (const float*)d_in[18];
  const float* gcn_b2  = (const float*)d_in[19];
  const float* gcn_cw  = (const float*)d_in[20];
  const float* gcn_cb  = (const float*)d_in[21];

  const int* row = ei;
  const int* col = ei + EE;

  float* out_h    = (float*)d_out;                       // [N,O]
  float* out_emb  = out_h + (size_t)NN * OO;             // [N,D]
  float* out_hemb = out_emb + (size_t)NN * DD;           // [N,H]  (== xs[0])

  float* W = (float*)d_ws;
  const size_t o_sums = 0, o_scale = 128, o_shift = 192, o_coefs = 256,
               o_wsm = 272, o_biasH = 528, o_Mb = 640;
  const size_t o_Wswz = o_Mb + (size_t)KA * HH / 2;      // Mb bf16 = KA*HH/2 floats
  const size_t o_degR = o_Wswz + (HH * HH + HH * OO + DD * DD) / 2;
  const size_t o_degC = o_degR + NN;
  const size_t o_dis  = o_degC + NN;
  const size_t o_odn  = o_dis + NN;
  const size_t o_idn  = o_odn + NN;
  const size_t o_xs1  = o_idn + NN;
  const size_t o_xs2  = o_xs1 + (size_t)NN * HH;
  const size_t o_xs3  = o_xs2 + (size_t)NN * HH;
  const size_t o_tA   = o_xs3 + (size_t)NN * HH;
  __bf16* Mb   = reinterpret_cast<__bf16*>(W + o_Mb);
  __bf16* Wswz = reinterpret_cast<__bf16*>(W + o_Wswz);

  const int EWGRID = 12500;             // ceil(N*64/256)
  const int WGRID  = (NN + 31) / 32;    // 1563 wave-tile blocks

  // ---- stats & degrees
  hipMemsetAsync(W + o_sums, 0, 128 * sizeof(float), stream);
  hipMemsetAsync(W + o_degR, 0, 2 * (size_t)NN * sizeof(float), stream);
  k_bn_stats<<<512, 256, 0, stream>>>(in_feat, W + o_sums);
  k_bn_final<<<1, 64, 0, stream>>>(W + o_sums, bn_g, bn_b, W + o_scale, W + o_shift);
  k_deg<<<1024, 256, 0, stream>>>(row, col, W + o_degR, W + o_degC);
  k_norms<<<(NN + 255) / 256, 256, 0, stream>>>(W + o_degR, W + o_degC,
                                                W + o_dis, W + o_odn, W + o_idn);
  // ---- FSTNN pre-contraction + weight swizzles + scalars
  k_precM<<<KA, 64, 0, stream>>>(fst_w2, fst_cw, Mb);
  k_biasH<<<1, 64, 0, stream>>>(fst_b2, fst_cw, fst_cb, W + o_biasH);
  k_swzW<<<(HH * HH + HH * OO + DD * DD + 255) / 256, 256, 0, stream>>>(
      out_w1, out_w2, gcn_cw, Wswz);
  k_scalars<<<1, 64, 0, stream>>>(alphas, combw, W + o_coefs, W + o_wsm);

  // ---- h_emb (xs0) via WMMA
  k_hemb<<<WGRID, 64, 0, stream>>>(in_feat, W + o_scale, W + o_shift,
                                   fst_w1, fst_b1, Mb, W + o_biasH, out_hemb);

  // ---- Jacobi recurrence
  hipMemsetAsync(W + o_tA, 0, (size_t)NN * HH * sizeof(float), stream);
  k_spmm_sym<<<2048, 256, 0, stream>>>(row, col, W + o_dis, out_hemb, W + o_tA);
  k_jstep1<<<EWGRID, 256, 0, stream>>>(out_hemb, W + o_tA, W + o_coefs, W + o_xs1);

  hipMemsetAsync(W + o_tA, 0, (size_t)NN * HH * sizeof(float), stream);
  k_spmm_sym<<<2048, 256, 0, stream>>>(row, col, W + o_dis, W + o_xs1, W + o_tA);
  k_jstepL<<<EWGRID, 256, 0, stream>>>(W + o_tA, W + o_xs1, out_hemb,
                                       W + o_coefs, 2, W + o_xs2);

  hipMemsetAsync(W + o_tA, 0, (size_t)NN * HH * sizeof(float), stream);
  k_spmm_sym<<<2048, 256, 0, stream>>>(row, col, W + o_dis, W + o_xs2, W + o_tA);
  k_jstepL<<<EWGRID, 256, 0, stream>>>(W + o_tA, W + o_xs2, W + o_xs1,
                                       W + o_coefs, 5, W + o_xs3);

  // ---- combination + output MLP (hc reuses tA)
  k_comb<<<EWGRID, 256, 0, stream>>>(out_hemb, W + o_xs1, W + o_xs2, W + o_xs3,
                                     W + o_wsm, W + o_tA);
  k_mlp<<<WGRID, 64, 0, stream>>>(W + o_tA, Wswz, out_b1, out_b2, out_h);

  // ---- DimensionWiseGCN branch (reuses xs1/xs2/xs3)
  hipMemsetAsync(W + o_xs1, 0, (size_t)NN * DD * sizeof(float), stream);
  k_spmm_gcn<<<2048, 256, 0, stream>>>(row, col, W + o_odn, W + o_idn, out_hemb, W + o_xs1);
  k_g2<<<EWGRID, 256, 0, stream>>>(W + o_xs1, gcn_w1, gcn_b1, gcn_w2, W + o_xs2);
  hipMemsetAsync(W + o_xs3, 0, (size_t)NN * DD * sizeof(float), stream);
  k_spmm_gcn<<<2048, 256, 0, stream>>>(row, col, W + o_odn, W + o_idn, W + o_xs2, W + o_xs3);
  k_gcn_out<<<WGRID, 64, 0, stream>>>(W + o_xs3, gcn_b2, Wswz + HH * HH + HH * OO,
                                      gcn_cb, out_emb);
}
